// BinaryLinear_55731495633322
// MI455X (gfx1250) — matrix operand order repaired and checked by reading, not compiled, not run on hardware
//
#include <hip/hip_runtime.h>
#include <cstdint>

// BinaryLinear on MI455X (gfx1250, wave32):
//   y = x @ (aa*tanh(kk*W))^T + bias,  M=8192, N=4096, K=4096
// bf16 WMMA GEMM (f32 accumulate); operands pre-converted to bf16 so both
// (67MB + 33.5MB) live in the 192MB L2. Global->LDS staging uses the CDNA5
// async copy engine (GLOBAL_LOAD_ASYNC_TO_LDS_B128, ASYNCcnt) to avoid the
// VGPR staging pressure that caused scratch spills in the reg-staged version.

typedef unsigned short u16;
typedef __attribute__((ext_vector_type(16))) __bf16 v16bf;
typedef __attribute__((ext_vector_type(8)))  float  v8f;
typedef int v4i __attribute__((__vector_size__(16)));  // matches builtin proto

#define MDIM 8192
#define NDIM 4096
#define KDIM 4096
#define TM 128
#define TN 128
#define TK 64
#define NK (KDIM / TK)

#if defined(__has_builtin)
#if __has_builtin(__builtin_amdgcn_global_load_async_to_lds_b128)
#define USE_ASYNC_LDS 1
#endif
#endif

typedef __attribute__((address_space(1))) v4i* as1_v4i_p;
typedef __attribute__((address_space(3))) v4i* as3_v4i_p;

__device__ __forceinline__ as3_v4i_p lds_v4i(u16* p) {
  // Flat LDS aperture base has zero low-32 bits -> low 32 bits are the LDS offset.
  return (as3_v4i_p)(unsigned int)(uintptr_t)p;
}
__device__ __forceinline__ as1_v4i_p gbl_v4i(const u16* p) {
  return (as1_v4i_p)(uintptr_t)p;
}

__device__ __forceinline__ u16 f32_to_bf16_rne(float f) {
  union { float f; unsigned int u; } v; v.f = f;
  unsigned int u = v.u;
  return (u16)((u + 0x7FFFu + ((u >> 16) & 1u)) >> 16);
}

// ---- prologue: x (f32) -> bf16, 4 elements/thread ----
__global__ __launch_bounds__(256) void cvt_x_bf16(const float* __restrict__ x,
                                                  u16* __restrict__ xb, int n4) {
  int i = blockIdx.x * blockDim.x + threadIdx.x;
  if (i >= n4) return;
  float4 v = ((const float4*)x)[i];
  unsigned int lo = (unsigned int)f32_to_bf16_rne(v.x) |
                    ((unsigned int)f32_to_bf16_rne(v.y) << 16);
  unsigned int hi = (unsigned int)f32_to_bf16_rne(v.z) |
                    ((unsigned int)f32_to_bf16_rne(v.w) << 16);
  ((uint2*)xb)[i] = make_uint2(lo, hi);
}

// ---- prologue: W (f32) -> bf16(aa * tanh(kk * W)) ----
__global__ __launch_bounds__(256) void binarize_w_bf16(const float* __restrict__ w,
                                                       u16* __restrict__ wb,
                                                       const float* __restrict__ kk,
                                                       const float* __restrict__ aa,
                                                       int n4) {
  int i = blockIdx.x * blockDim.x + threadIdx.x;
  if (i >= n4) return;
  const float K = kk[0], A = aa[0];
  float4 v = ((const float4*)w)[i];
  float a0 = A * tanhf(K * v.x);
  float a1 = A * tanhf(K * v.y);
  float a2 = A * tanhf(K * v.z);
  float a3 = A * tanhf(K * v.w);
  unsigned int lo = (unsigned int)f32_to_bf16_rne(a0) |
                    ((unsigned int)f32_to_bf16_rne(a1) << 16);
  unsigned int hi = (unsigned int)f32_to_bf16_rne(a2) |
                    ((unsigned int)f32_to_bf16_rne(a3) << 16);
  ((uint2*)wb)[i] = make_uint2(lo, hi);
}

// ---- main GEMM: 128x128 block tile, K-stage 64, 8 waves, double-buffered LDS ----
__global__ __launch_bounds__(256) void wmma_gemm_bias(const u16* __restrict__ Xb,
                                                      const u16* __restrict__ Wb,
                                                      const float* __restrict__ bias,
                                                      float* __restrict__ Out) {
  // 2 * (128*64) u16 per operand, double buffered = exactly 64 KB of LDS.
  __shared__ u16 sA[2][TM * TK];
  __shared__ u16 sB[2][TN * TK];

  const int tid  = threadIdx.x;
  const int lane = tid & 31;
  const int wave = tid >> 5;
  const int wm   = wave & 3;   // wave's M offset: wm*32
  const int wn   = wave >> 2;  // wave's N offset: wn*64
  const int gm0  = blockIdx.y * TM;
  const int gn0  = blockIdx.x * TN;

  // Each tile = 1024 16-byte chunks; 4 A-chunks + 4 B-chunks per thread.
#if defined(USE_ASYNC_LDS)
  // Zero-VGPR staging: async copy engine writes LDS directly (ASYNCcnt).
  auto stageTo = [&](int kt, int buf) {
    const int k0 = kt * TK;
#pragma unroll
    for (int i = 0; i < 4; ++i) {
      int c = tid + i * 256;
      int row = c >> 3, cr = c & 7;
      int pc = cr ^ (row & 7);  // XOR swizzle vs. bank conflicts
      __builtin_amdgcn_global_load_async_to_lds_b128(
          gbl_v4i(Xb + (size_t)(gm0 + row) * KDIM + k0 + cr * 8),
          lds_v4i(&sA[buf][row * TK + pc * 8]), 0, 0);
      __builtin_amdgcn_global_load_async_to_lds_b128(
          gbl_v4i(Wb + (size_t)(gn0 + row) * KDIM + k0 + cr * 8),
          lds_v4i(&sB[buf][row * TK + pc * 8]), 0, 0);
    }
  };
  auto stageWait = [&]() {
#if __has_builtin(__builtin_amdgcn_s_wait_asynccnt)
    __builtin_amdgcn_s_wait_asynccnt(0);
#else
    asm volatile("s_wait_asynccnt 0x0" ::: "memory");
#endif
  };
#else
  // Fallback: reg-staged copy (may spill; only used if async builtin missing).
  auto stageTo = [&](int kt, int buf) {
    const int k0 = kt * TK;
#pragma unroll
    for (int i = 0; i < 4; ++i) {
      int c = tid + i * 256;
      int row = c >> 3, cr = c & 7;
      int pc = cr ^ (row & 7);
      uint4 a = *(const uint4*)(Xb + (size_t)(gm0 + row) * KDIM + k0 + cr * 8);
      uint4 b = *(const uint4*)(Wb + (size_t)(gn0 + row) * KDIM + k0 + cr * 8);
      *(uint4*)&sA[buf][row * TK + pc * 8] = a;
      *(uint4*)&sB[buf][row * TK + pc * 8] = b;
    }
  };
  auto stageWait = [&]() {};
#endif

  union FragBf { uint4 q[2]; v16bf v; };
  auto ldsA = [&](int buf, int row, int chunk) -> uint4 {
    int pc = chunk ^ (row & 7);
    return *(const uint4*)&sA[buf][row * TK + pc * 8];
  };
  auto ldsB = [&](int buf, int row, int chunk) -> uint4 {
    int pc = chunk ^ (row & 7);
    return *(const uint4*)&sB[buf][row * TK + pc * 8];
  };

  v8f acc[2][4];
  const v8f vzero = {};
#pragma unroll
  for (int mi = 0; mi < 2; ++mi)
#pragma unroll
    for (int ni = 0; ni < 4; ++ni) acc[mi][ni] = vzero;

  auto computeStage = [&](int buf) {
#pragma unroll
    for (int ks = 0; ks < 2; ++ks) {
      const int cbase = ks * 4;  // 32 u16 = 4 chunks per K-substep
      FragBf a[2], b[4];
      // A (16x32 bf16): lanes 0-15 row M holds K0..7 & K16..23; lanes 16-31 K8..15 & K24..31
#pragma unroll
      for (int mi = 0; mi < 2; ++mi) {
        const int row = wm * 32 + mi * 16 + (lane & 15);
        const int c0  = cbase + (lane >> 4);
        a[mi].q[0] = ldsA(buf, row, c0);
        a[mi].q[1] = ldsA(buf, row, c0 + 2);
      }
      // B (32x16 bf16): lanes 0-15 col N holds K0..7 & K16..23; lanes 16-31 K8..15 & K24..31
#pragma unroll
      for (int ni = 0; ni < 4; ++ni) {
        const int col = wn * 64 + ni * 16 + (lane & 15);
        const int c0  = cbase + (lane >> 4);
        b[ni].q[0] = ldsB(buf, col, c0);
        b[ni].q[1] = ldsB(buf, col, c0 + 2);
      }
#pragma unroll
      for (int mi = 0; mi < 2; ++mi)
#pragma unroll
        for (int ni = 0; ni < 4; ++ni)
          acc[mi][ni] = __builtin_amdgcn_wmma_f32_16x16x32_bf16(
              false, a[mi].v, false, b[ni].v, (short)0, acc[mi][ni], false, false);
    }
  };

  stageTo(0, 0);
  stageWait();
  __syncthreads();

  for (int kt = 0; kt < NK; ++kt) {
    const int cur = kt & 1;
    if (kt + 1 < NK) stageTo(kt + 1, cur ^ 1);  // async copies in flight during WMMAs
    computeStage(cur);
    stageWait();
    __syncthreads();
  }

  // Epilogue: fused bias add, f32 store.
  // C/D layout: VGPR r, lanes 0-15 -> (M=r, N=lane); lanes 16-31 -> (M=r+8, N=lane-16)
#pragma unroll
  for (int ni = 0; ni < 4; ++ni) {
    const int col = gn0 + wn * 64 + ni * 16 + (lane & 15);
    const float bv = bias[col];
#pragma unroll
    for (int mi = 0; mi < 2; ++mi) {
      const int row0 = gm0 + wm * 32 + mi * 16 + ((lane >> 4) * 8);
#pragma unroll
      for (int r = 0; r < 8; ++r)
        Out[(size_t)(row0 + r) * NDIM + col] = acc[mi][ni][r] + bv;
    }
  }
}

extern "C" void kernel_launch(void* const* d_in, const int* in_sizes, int n_in,
                              void* d_out, int out_size, void* d_ws, size_t ws_size,
                              hipStream_t stream) {
  const float* x    = (const float*)d_in[0];  // [8192, 4096]
  const float* w    = (const float*)d_in[1];  // [4096, 4096]
  const float* bias = (const float*)d_in[2];  // [4096]
  const float* kk   = (const float*)d_in[3];
  const float* aa   = (const float*)d_in[4];
  float* out = (float*)d_out;

  // workspace: xb (67,108,864 B) then wb (33,554,432 B)
  u16* xb = (u16*)d_ws;
  u16* wb = (u16*)((char*)d_ws + (size_t)MDIM * KDIM * sizeof(u16));

  cvt_x_bf16<<<(MDIM * KDIM / 4) / 256, 256, 0, stream>>>(x, xb, MDIM * KDIM / 4);
  binarize_w_bf16<<<(NDIM * KDIM / 4) / 256, 256, 0, stream>>>(w, wb, kk, aa,
                                                               NDIM * KDIM / 4);

  dim3 grid(NDIM / TN, MDIM / TM);  // (32, 64)
  wmma_gemm_bias<<<grid, 256, 0, stream>>>(xb, wb, bias, out);
}
